// CBAMPatchMambaAttAVPPrototype_49701361549633
// MI455X (gfx1250) — matrix operand order repaired and checked by reading, not compiled, not run on hardware
//
#include <hip/hip_runtime.h>
#include <hip/hip_bf16.h>
#include <math.h>

#define BATCH   16
#define C_IN    6
#define T_IN    4096
#define PATCH   16
#define SEQ     256          /* T_IN / PATCH */
#define D_MODEL 256
#define D_INNER 512
#define GROUPS  8
#define GSIZE   64
#define DSTATE  16
#define RAWN    264          /* GROUPS*(1+2*DSTATE) */
#define RAWP    288          /* RAWN padded to 32-col multiple for full WMMA tiles */
#define N_HEADS 4
#define HD      64
#define NCLS    8
#define BNF_C   0.9999950000374997f

typedef __attribute__((ext_vector_type(16))) _Float16 v16h;
typedef __attribute__((ext_vector_type(8)))  float    v8f;

__device__ __forceinline__ float siluf(float x) { return x / (1.0f + __expf(-x)); }
__device__ __forceinline__ float sigmf(float x) { return 1.0f / (1.0f + __expf(-x)); }
__device__ __forceinline__ float splusf(float x) { return (x > 20.0f) ? x : log1pf(__expf(x)); }

// ---------------------------------------------------------------------------
// Pack B (f32, row-major KxN, or transposed NxK if transb) into this code's
// 32x16-f16 B-fragment layout, zero-filling padded columns. One thread emits
// one lane's v16h fragment (32B store). bp[( (nt*K32 + ks)*32 + lane )*16 + e]
// holds B[ks*32 + (lane>>4)*16 + e][nt*16 + (lane&15)].
// ---------------------------------------------------------------------------
__global__ void k_pack_b(const float* __restrict__ Bm, int ldb, _Float16* __restrict__ bp,
                         int N, int K, int transb, int total)
{
    const int id = blockIdx.x * blockDim.x + threadIdx.x;
    if (id >= total) return;
    const int K32  = K >> 5;
    const int lane = id & 31;
    const int ks   = (id >> 5) % K32;
    const int nt   = (id >> 5) / K32;
    const int col  = (nt << 4) + (lane & 15);
    const int kb   = (ks << 5) + ((lane >> 4) << 4);
    v16h frag;
#pragma unroll
    for (int e = 0; e < 16; ++e) {
        float v = 0.0f;
        if (col < N)
            v = transb ? Bm[(size_t)col * ldb + kb + e]
                       : Bm[(size_t)(kb + e) * ldb + col];
        frag[e] = (_Float16)v;
    }
    *(v16h*)(bp + (size_t)id * 16) = frag;
}

// ---------------------------------------------------------------------------
// WMMA GEMM: C[M, Npad] (+)= act( A[M,K](f32) * Bpacked + bias ).
// M % 16 == 0, K % 32 == 0, N padded to 32 (nnt2 = padded_N/32).
// One wave owns a 16x32 C strip (2 WMMA tiles sharing the A fragment).
// No data-dependent branches: EXEC stays all-ones for every WMMA.
// ACT: 0=none, 1=silu.  ACC: accumulate into existing C (residual add).
// ---------------------------------------------------------------------------
template <int ACT, bool ACC>
__global__ __launch_bounds__(256) void gemm_wmma(
    const float* __restrict__ A, int lda,
    const _Float16* __restrict__ Bp,
    const float* __restrict__ bias,
    float* __restrict__ C, int ldc,
    int M, int nnt2, int K)
{
    const int lane = threadIdx.x & 31;
    const int wave = threadIdx.x >> 5;
    const int half = lane >> 4;
    const int r    = lane & 15;
    const int K32  = K >> 5;
    const int tile = blockIdx.x * 8 + wave;
    if (tile >= (M >> 4) * nnt2) return;           // wave-uniform
    const int m0  = (tile / nnt2) << 4;
    const int nt0 = (tile % nnt2) << 1;

    const float*    ap0 = A  + (size_t)(m0 + r) * lda + (half << 4);
    const _Float16* bp0 = Bp + ((size_t)(nt0    ) * K32 * 32 + lane) * 16;
    const _Float16* bp1 = Bp + ((size_t)(nt0 + 1) * K32 * 32 + lane) * 16;

    v8f acc0 = {}, acc1 = {};
    for (int ks = 0; ks < K32; ++ks) {
        const float* ap = ap0 + (ks << 5);
        const float4 a0 = *(const float4*)(ap);
        const float4 a1 = *(const float4*)(ap + 4);
        const float4 a2 = *(const float4*)(ap + 8);
        const float4 a3 = *(const float4*)(ap + 12);
        v16h af;
        af[0]  = (_Float16)a0.x; af[1]  = (_Float16)a0.y;
        af[2]  = (_Float16)a0.z; af[3]  = (_Float16)a0.w;
        af[4]  = (_Float16)a1.x; af[5]  = (_Float16)a1.y;
        af[6]  = (_Float16)a1.z; af[7]  = (_Float16)a1.w;
        af[8]  = (_Float16)a2.x; af[9]  = (_Float16)a2.y;
        af[10] = (_Float16)a2.z; af[11] = (_Float16)a2.w;
        af[12] = (_Float16)a3.x; af[13] = (_Float16)a3.y;
        af[14] = (_Float16)a3.z; af[15] = (_Float16)a3.w;
        const v16h b0 = *(const v16h*)(bp0 + (size_t)ks * 512);
        const v16h b1 = *(const v16h*)(bp1 + (size_t)ks * 512);
        acc0 = __builtin_amdgcn_wmma_f32_16x16x32_f16(
            false, af, false, b0, (short)0, acc0, false, false);
        acc1 = __builtin_amdgcn_wmma_f32_16x16x32_f16(
            false, af, false, b1, (short)0, acc1, false, false);
    }

    const int col0 = (nt0 << 4) + r;
#pragma unroll
    for (int v = 0; v < 8; ++v) {
        const int row = m0 + (half << 3) + v;
        const size_t i0 = (size_t)row * ldc + col0;
        float t0 = acc0[v], t1 = acc1[v];
        if (bias) { t0 += bias[col0]; t1 += bias[col0 + 16]; }
        if (ACT == 1) { t0 = siluf(t0); t1 = siluf(t1); }
        if (ACC) { t0 += C[i0]; t1 += C[i0 + 16]; }
        C[i0] = t0; C[i0 + 16] = t1;
    }
}

// ---------------------------------------------------------------------------
// CBAM stage
// ---------------------------------------------------------------------------
__global__ void k_meanmax(const float* __restrict__ x, float* meanv, float* maxv)
{
    __shared__ float ssum[256], smax[256];
    const int bc = blockIdx.x;
    const float* xr = x + (size_t)bc * T_IN;
    float s = 0.0f, m = -3.4e38f;
    for (int t = threadIdx.x; t < T_IN; t += 256) {
        const float v = xr[t];
        s += v; m = fmaxf(m, v);
    }
    ssum[threadIdx.x] = s; smax[threadIdx.x] = m;
    __syncthreads();
    for (int st = 128; st > 0; st >>= 1) {
        if (threadIdx.x < st) {
            ssum[threadIdx.x] += ssum[threadIdx.x + st];
            smax[threadIdx.x] = fmaxf(smax[threadIdx.x], smax[threadIdx.x + st]);
        }
        __syncthreads();
    }
    if (threadIdx.x == 0) {
        meanv[bc] = ssum[0] * (1.0f / T_IN);
        maxv[bc]  = smax[0];
    }
}

__global__ void k_cbam_mlp(const float* meanv, const float* maxv,
                           const float* w1, const float* b1,
                           const float* w2, const float* b2,
                           const float* w3, const float* b3, float* ca)
{
    const int b = threadIdx.x;
    if (b >= BATCH) return;
    float o[2][C_IN];
    for (int pass = 0; pass < 2; ++pass) {
        const float* vv = (pass ? maxv : meanv) + b * C_IN;
        float h1[8], h2[8];
        for (int j = 0; j < 8; ++j) {
            float s = b1[j];
            for (int c = 0; c < C_IN; ++c) s += vv[c] * w1[c * 8 + j];
            h1[j] = fmaxf(s * BNF_C, 0.0f);
        }
        for (int j = 0; j < 8; ++j) {
            float s = b2[j];
            for (int i = 0; i < 8; ++i) s += h1[i] * w2[i * 8 + j];
            h2[j] = fmaxf(s * BNF_C, 0.0f);
        }
        for (int c = 0; c < C_IN; ++c) {
            float s = b3[c];
            for (int i = 0; i < 8; ++i) s += h2[i] * w3[i * C_IN + c];
            o[pass][c] = s;
        }
    }
    for (int c = 0; c < C_IN; ++c) ca[b * C_IN + c] = sigmf(o[0][c] + o[1][c]);
}

__global__ void k_chanscale(const float* __restrict__ x, const float* __restrict__ ca,
                            float* __restrict__ out1)
{
    const int id = blockIdx.x * blockDim.x + threadIdx.x;
    out1[id] = ca[id / T_IN] * x[id];
}

__global__ void k_spool(const float* __restrict__ out1, float* __restrict__ sp)
{
    const int id = blockIdx.x * blockDim.x + threadIdx.x;
    const int b = id / T_IN, t = id % T_IN;
    float s = 0.0f, m = -3.4e38f;
    for (int c = 0; c < C_IN; ++c) {
        const float v = out1[((size_t)b * C_IN + c) * T_IN + t];
        s += v; m = fmaxf(m, v);
    }
    sp[((size_t)b * 2 + 0) * T_IN + t] = s * (1.0f / C_IN);
    sp[((size_t)b * 2 + 1) * T_IN + t] = m;
}

__global__ void k_conv1d7(const float* __restrict__ in, const float* __restrict__ w,
                          float* __restrict__ out, int Cin, int Cout, int act, int n)
{
    const int id = blockIdx.x * blockDim.x + threadIdx.x;
    if (id >= n) return;
    const int t  = id % T_IN;
    const int oc = (id / T_IN) % Cout;
    const int b  = id / (T_IN * Cout);
    float s = 0.0f;
    for (int ic = 0; ic < Cin; ++ic)
        for (int kk = 0; kk < 7; ++kk) {
            const int tt = t + kk - 3;
            if (tt >= 0 && tt < T_IN)
                s += in[((size_t)b * Cin + ic) * T_IN + tt] * w[(oc * Cin + ic) * 7 + kk];
        }
    s *= BNF_C;
    out[id] = act ? sigmf(s) : fmaxf(s, 0.0f);
}

__global__ void k_cbam_combine(const float* __restrict__ x, const float* __restrict__ out1,
                               const float* __restrict__ att, const float* __restrict__ rw,
                               float* __restrict__ y)
{
    const int id = blockIdx.x * blockDim.x + threadIdx.x;
    const int t = id % T_IN;
    const int b = id / (C_IN * T_IN);
    y[id] = att[b * T_IN + t] * out1[id] + x[id] * rw[0];
}

// (B,C,T_IN) -> (B*SEQ, C_IN*PATCH) im2col for the patch-embed GEMM
__global__ void k_im2col(const float* __restrict__ xin, float* __restrict__ patches)
{
    const int id = blockIdx.x * blockDim.x + threadIdx.x;
    const int col = id % (C_IN * PATCH);
    const int r   = (id / (C_IN * PATCH)) % SEQ;
    const int b   = id / (C_IN * PATCH * SEQ);
    const int c = col >> 4, kk = col & 15;
    patches[id] = xin[((size_t)b * C_IN + c) * T_IN + r * PATCH + kk];
}

__global__ void k_rmsnorm(const float* __restrict__ x, const float* __restrict__ g,
                          float* __restrict__ out)
{
    const int row = blockIdx.x * blockDim.x + threadIdx.x;
    const float* xr = x + (size_t)row * D_MODEL;
    float ss = 0.0f;
    for (int i = 0; i < D_MODEL; ++i) ss += xr[i] * xr[i];
    const float inv = rsqrtf(ss * (1.0f / D_MODEL) + 1e-6f);
    float* orow = out + (size_t)row * D_MODEL;
    for (int i = 0; i < D_MODEL; ++i) orow[i] = xr[i] * inv * g[i];
}

__global__ void k_silu_xz(float* __restrict__ xzg)
{
    const int id = blockIdx.x * blockDim.x + threadIdx.x;
    const int row = id >> 9, d = id & (D_INNER - 1);
    const size_t idx = (size_t)row * (2 * D_INNER) + d;
    xzg[idx] = siluf(xzg[idx]);
}

// ---------------------------------------------------------------------------
// Selective scan: sequential over SEQ per (batch, channel); dir=1 is the
// time-reversed scan (== flip/scan/flip of the reference).
// raw row stride RAWP; group g layout [delta, B[16], C[16]] at g*33.
// ---------------------------------------------------------------------------
__global__ void k_ssm_scan(const float* __restrict__ xzg, const float* __restrict__ raw,
                           const float* __restrict__ A_log, float* __restrict__ ydir, int dir)
{
    const int id = blockIdx.x * blockDim.x + threadIdx.x;
    const int b = id / D_INNER, d = id % D_INNER;
    const int g = d / GSIZE;
    float A[DSTATE];
#pragma unroll
    for (int n = 0; n < DSTATE; ++n) A[n] = -__expf(A_log[n]);
    float h[DSTATE] = {};
    for (int s = 0; s < SEQ; ++s) {
        const int t = dir ? (SEQ - 1 - s) : s;
        const int row = b * SEQ + t;
        const float* rr = raw + (size_t)row * RAWP + g * 33;
        const float delta = splusf(rr[0]);
        const float u = xzg[(size_t)row * (2 * D_INNER) + d];
        float y = 0.0f;
#pragma unroll
        for (int n = 0; n < DSTATE; ++n) {
            float dA = delta * A[n];
            dA = fminf(10.0f, fmaxf(-10.0f, dA));
            const float eA = __expf(dA);
            const float frac = (fabsf(dA) < 1e-4f) ? delta : (eA - 1.0f) / (A[n] + 1e-12f);
            h[n] = h[n] * eA + frac * rr[1 + n] * u;
            y += h[n] * rr[17 + n];
        }
        ydir[(size_t)row * D_INNER + d] = y;
    }
}

__global__ void k_combine_gate(float* __restrict__ yf, const float* __restrict__ yb,
                               const float* __restrict__ xzg)
{
    const int id = blockIdx.x * blockDim.x + threadIdx.x;
    const int row = id / D_INNER, d = id % D_INNER;
    const float gate = xzg[(size_t)row * (2 * D_INNER) + D_INNER + d];
    yf[id] = 0.5f * (yf[id] + yb[id]) * siluf(gate);
}

// ---------------------------------------------------------------------------
// Attention pooling + logits
// ---------------------------------------------------------------------------
__global__ void k_qproj(const float* __restrict__ cls, const float* __restrict__ qw,
                        const float* __restrict__ qb, float* __restrict__ q)
{
    const int n = threadIdx.x;
    float s = qb[n];
    for (int dd = 0; dd < D_MODEL; ++dd) s += cls[dd] * qw[dd * D_MODEL + n];
    q[n] = s;
}

__global__ void k_attn_pool(const float* __restrict__ q, const float* __restrict__ k,
                            const float* __restrict__ v, float* __restrict__ out)
{
    __shared__ float sc[SEQ];
    __shared__ float red[256];
    const int b = blockIdx.x / N_HEADS, h = blockIdx.x % N_HEADS;
    const int j = threadIdx.x;
    const float* qh = q + h * HD;
    const float* kr = k + ((size_t)(b * SEQ + j)) * D_MODEL + h * HD;
    float dot = 0.0f;
    for (int dd = 0; dd < HD; ++dd) dot += qh[dd] * kr[dd];
    sc[j] = dot * 0.125f;
    red[j] = sc[j];
    __syncthreads();
    for (int st = 128; st > 0; st >>= 1) {
        if (j < st) red[j] = fmaxf(red[j], red[j + st]);
        __syncthreads();
    }
    const float mx = red[0];
    __syncthreads();
    const float e = __expf(sc[j] - mx);
    sc[j] = e; red[j] = e;
    __syncthreads();
    for (int st = 128; st > 0; st >>= 1) {
        if (j < st) red[j] += red[j + st];
        __syncthreads();
    }
    const float inv = 1.0f / red[0];
    __syncthreads();
    sc[j] *= inv;
    __syncthreads();
    if (j < HD) {
        float acc = 0.0f;
        for (int jj = 0; jj < SEQ; ++jj)
            acc += sc[jj] * v[((size_t)(b * SEQ + jj)) * D_MODEL + h * HD + j];
        out[b * D_MODEL + h * HD + j] = acc;
    }
}

__global__ void k_logits(const float* __restrict__ z, const float* __restrict__ proto,
                         float* __restrict__ out)
{
    const int id = threadIdx.x;
    const int b = id / NCLS, c = id % NCLS;
    float s = 0.0f;
    for (int i = 0; i < D_MODEL; ++i) {
        const float d = z[b * D_MODEL + i] - proto[c * D_MODEL + i];
        s += d * d;
    }
    out[id] = -s;
}

// ---------------------------------------------------------------------------
static inline int nnt2_of(int N) { return (N + 31) / 32; }
static inline int gemm_grid(int M, int N)
{
    return ((M >> 4) * nnt2_of(N) + 7) / 8;       // 8 waves / 256-thread block
}

extern "C" void kernel_launch(void* const* d_in, const int* in_sizes, int n_in,
                              void* d_out, int out_size, void* d_ws, size_t ws_size,
                              hipStream_t stream)
{
    (void)in_sizes; (void)n_in; (void)out_size; (void)ws_size;
    auto F = [&](int i) -> const float* { return (const float*)d_in[i]; };

    const float* x = F(0);
    const float *cw1 = F(1), *cb1 = F(2), *cw2 = F(3), *cb2 = F(4), *cw3 = F(5), *cb3 = F(6);
    const float *conv1 = F(7), *conv2 = F(8), *conv3 = F(9), *rw = F(10);
    const float *pw = F(11), *pb = F(12);
    const float *cls = F(53), *qw = F(54), *qb = F(55), *kw = F(56), *kb = F(57);
    const float *vw = F(58), *vb = F(59), *ow = F(60), *ob = F(61);
    const float* proto = F(62);
    float* logits = (float*)d_out;

    float* wsf = (float*)d_ws;
    size_t off = 0;
    auto alloc = [&](size_t n) { off = (off + 7) & ~(size_t)7; float* p = wsf + off; off += n; return p; };
    auto allocH = [&](size_t nh) { return (_Float16*)alloc((nh + 1) / 2); };

    float* meanv  = alloc(BATCH * C_IN);
    float* maxv   = alloc(BATCH * C_IN);
    float* ca     = alloc(BATCH * C_IN);
    float* out1   = alloc((size_t)BATCH * C_IN * T_IN);
    float* sp     = alloc((size_t)BATCH * 2 * T_IN);
    float* a1     = alloc((size_t)BATCH * 8 * T_IN);
    float* a2     = alloc((size_t)BATCH * 8 * T_IN);
    float* att    = alloc((size_t)BATCH * T_IN);
    float* xcb    = alloc((size_t)BATCH * C_IN * T_IN);
    float* patch  = alloc((size_t)BATCH * SEQ * C_IN * PATCH);
    float* xtok   = alloc((size_t)BATCH * SEQ * D_MODEL);
    float* zbuf   = alloc((size_t)BATCH * SEQ * D_MODEL);
    float* xzg    = alloc((size_t)BATCH * SEQ * 2 * D_INNER);
    float* raw    = alloc((size_t)BATCH * SEQ * RAWP);
    float* yf     = alloc((size_t)BATCH * SEQ * D_INNER);
    float* yb     = alloc((size_t)BATCH * SEQ * D_INNER);
    float* kbuf   = alloc((size_t)BATCH * SEQ * D_MODEL);
    float* vbuf   = alloc((size_t)BATCH * SEQ * D_MODEL);
    float* qv     = alloc(D_MODEL);
    float* pooled = alloc(BATCH * D_MODEL);
    float* zf     = alloc(BATCH * D_MODEL);
    _Float16* pk_patch = allocH((size_t)256  * 96);
    _Float16* pk_in    = allocH((size_t)1024 * 256);
    _Float16* pk_pp    = allocH((size_t)RAWP * 512);
    _Float16* pk_out   = allocH((size_t)256  * 512);
    _Float16* pk_f1    = allocH((size_t)1024 * 256);
    _Float16* pk_f2    = allocH((size_t)256  * 1024);
    _Float16* pk_k     = allocH((size_t)256  * 256);
    _Float16* pk_v     = allocH((size_t)256  * 256);
    _Float16* pk_o     = allocH((size_t)256  * 256);

    auto packb = [&](const float* Bm, int ldb, int N, int K, int transb, _Float16* dst) {
        const int total = nnt2_of(N) * 2 * (K >> 5) * 32;
        k_pack_b<<<(total + 255) / 256, 256, 0, stream>>>(Bm, ldb, dst, N, K, transb, total);
    };

    const int R = BATCH * SEQ;                    // 4096 token rows

    // CBAM
    k_meanmax<<<BATCH * C_IN, 256, 0, stream>>>(x, meanv, maxv);
    k_cbam_mlp<<<1, 32, 0, stream>>>(meanv, maxv, cw1, cb1, cw2, cb2, cw3, cb3, ca);
    k_chanscale<<<(BATCH * C_IN * T_IN) / 256, 256, 0, stream>>>(x, ca, out1);
    k_spool<<<(BATCH * T_IN) / 256, 256, 0, stream>>>(out1, sp);
    k_conv1d7<<<(BATCH * 8 * T_IN) / 256, 256, 0, stream>>>(sp, conv1, a1, 2, 8, 0, BATCH * 8 * T_IN);
    k_conv1d7<<<(BATCH * 8 * T_IN) / 256, 256, 0, stream>>>(a1, conv2, a2, 8, 8, 0, BATCH * 8 * T_IN);
    k_conv1d7<<<(BATCH * T_IN) / 256, 256, 0, stream>>>(a2, conv3, att, 8, 1, 1, BATCH * T_IN);
    k_cbam_combine<<<(BATCH * C_IN * T_IN) / 256, 256, 0, stream>>>(x, out1, att, rw, xcb);

    // Patch embed (weight (256,96) used transposed)
    k_im2col<<<(BATCH * SEQ * C_IN * PATCH) / 256, 256, 0, stream>>>(xcb, patch);
    packb(pw, C_IN * PATCH, D_MODEL, C_IN * PATCH, 1, pk_patch);
    gemm_wmma<0, false><<<gemm_grid(R, D_MODEL), 256, 0, stream>>>(
        patch, C_IN * PATCH, pk_patch, pb, xtok, D_MODEL, R, nnt2_of(D_MODEL), C_IN * PATCH);

    // Mamba layers
    for (int L = 0; L < 4; ++L) {
        const float* n1g  = F(13 + L * 10 + 0);
        const float* inw  = F(13 + L * 10 + 1);
        const float* ppw  = F(13 + L * 10 + 2);
        const float* alog = F(13 + L * 10 + 3);
        const float* outw = F(13 + L * 10 + 4);
        const float* n2g  = F(13 + L * 10 + 5);
        const float* fw1  = F(13 + L * 10 + 6);
        const float* fb1  = F(13 + L * 10 + 7);
        const float* fw2  = F(13 + L * 10 + 8);
        const float* fb2  = F(13 + L * 10 + 9);

        packb(inw, 2 * D_INNER, 2 * D_INNER, D_MODEL, 0, pk_in);
        packb(ppw, RAWN, RAWN, D_INNER, 0, pk_pp);
        packb(outw, D_MODEL, D_MODEL, D_INNER, 0, pk_out);
        packb(fw1, 4 * D_MODEL, 4 * D_MODEL, D_MODEL, 0, pk_f1);
        packb(fw2, D_MODEL, D_MODEL, 4 * D_MODEL, 0, pk_f2);

        k_rmsnorm<<<R / 256, 256, 0, stream>>>(xtok, n1g, zbuf);
        gemm_wmma<0, false><<<gemm_grid(R, 2 * D_INNER), 256, 0, stream>>>(
            zbuf, D_MODEL, pk_in, nullptr, xzg, 2 * D_INNER, R, nnt2_of(2 * D_INNER), D_MODEL);
        k_silu_xz<<<(R * D_INNER) / 256, 256, 0, stream>>>(xzg);
        gemm_wmma<0, false><<<gemm_grid(R, RAWN), 256, 0, stream>>>(
            xzg, 2 * D_INNER, pk_pp, nullptr, raw, RAWP, R, nnt2_of(RAWN), D_INNER);
        k_ssm_scan<<<(BATCH * D_INNER) / 256, 256, 0, stream>>>(xzg, raw, alog, yf, 0);
        k_ssm_scan<<<(BATCH * D_INNER) / 256, 256, 0, stream>>>(xzg, raw, alog, yb, 1);
        k_combine_gate<<<(R * D_INNER) / 256, 256, 0, stream>>>(yf, yb, xzg);
        gemm_wmma<0, true><<<gemm_grid(R, D_MODEL), 256, 0, stream>>>(
            yf, D_INNER, pk_out, nullptr, xtok, D_MODEL, R, nnt2_of(D_MODEL), D_INNER);
        k_rmsnorm<<<R / 256, 256, 0, stream>>>(xtok, n2g, zbuf);
        gemm_wmma<1, false><<<gemm_grid(R, 4 * D_MODEL), 256, 0, stream>>>(
            zbuf, D_MODEL, pk_f1, fb1, xzg, 4 * D_MODEL, R, nnt2_of(4 * D_MODEL), D_MODEL);
        gemm_wmma<0, true><<<gemm_grid(R, D_MODEL), 256, 0, stream>>>(
            xzg, 4 * D_MODEL, pk_f2, fb2, xtok, D_MODEL, R, nnt2_of(D_MODEL), 4 * D_MODEL);
    }

    // Attention pooling + prototype logits
    packb(kw, D_MODEL, D_MODEL, D_MODEL, 0, pk_k);
    packb(vw, D_MODEL, D_MODEL, D_MODEL, 0, pk_v);
    packb(ow, D_MODEL, D_MODEL, D_MODEL, 0, pk_o);
    k_qproj<<<1, 256, 0, stream>>>(cls, qw, qb, qv);
    gemm_wmma<0, false><<<gemm_grid(R, D_MODEL), 256, 0, stream>>>(
        xtok, D_MODEL, pk_k, kb, kbuf, D_MODEL, R, nnt2_of(D_MODEL), D_MODEL);
    gemm_wmma<0, false><<<gemm_grid(R, D_MODEL), 256, 0, stream>>>(
        xtok, D_MODEL, pk_v, vb, vbuf, D_MODEL, R, nnt2_of(D_MODEL), D_MODEL);
    k_attn_pool<<<BATCH * N_HEADS, 256, 0, stream>>>(qv, kbuf, vbuf, pooled);
    gemm_wmma<0, false><<<gemm_grid(BATCH, D_MODEL), 256, 0, stream>>>(
        pooled, D_MODEL, pk_o, ob, zf, D_MODEL, BATCH, nnt2_of(D_MODEL), D_MODEL);
    k_logits<<<1, BATCH * NCLS, 0, stream>>>(zf, proto, logits);
}
